// SpMiddleFHDLite_48026324304215
// MI455X (gfx1250) — matrix operand order repaired and checked by reading, not compiled, not run on hardware
//
#include <hip/hip_runtime.h>
#include <hip/hip_bf16.h>

// ---------------------------------------------------------------------------
// SpMiddleFHDLite on gfx1250: densify (NDHWC4 f32) -> 4x implicit-GEMM conv3d
// via WMMA f16 + folded BN + ReLU + mask.  wave32, 16x16 output tiles/wave.
// Intermediates h1..h3 are f16 channels-last so each 8-wide K run of the A
// fragment is ONE 16-byte load.  Weights staged into LDS as f16 [co][Kpad]
// (K = tap*CinP + ci, zero padded) via the Tensor Data Mover when available.
// ---------------------------------------------------------------------------

typedef __attribute__((ext_vector_type(16))) _Float16 v16h;
typedef __attribute__((ext_vector_type(8)))  _Float16 v8h;
typedef __attribute__((ext_vector_type(8)))  float    v8f;
typedef __attribute__((ext_vector_type(4)))  unsigned int u32x4;
typedef __attribute__((ext_vector_type(8)))  int      i32x8;
typedef __attribute__((ext_vector_type(4)))  int      i32x4;

#if defined(__has_builtin)
#if __has_builtin(__builtin_amdgcn_tensor_load_to_lds)
#define HAVE_TDM 1
#endif
#endif

// ---------------- zeroing (float4 stores, n must be multiple of 4) ---------
__global__ void zero_kernel(float* __restrict__ p, long long n) {
    long long i = ((long long)blockIdx.x * blockDim.x + threadIdx.x) * 4;
    long long stride = (long long)gridDim.x * blockDim.x * 4;
    for (; i < n; i += stride) {
        *(float4*)(p + i) = make_float4(0.f, 0.f, 0.f, 0.f);
    }
}

// ---------------- voxel scatter into NDHWC4 dense (atomic add) -------------
__global__ void scatter_kernel(const float* __restrict__ vf,
                               const int* __restrict__ coors,
                               float* __restrict__ dense,   // (B,D,H,W,4)
                               float* __restrict__ mask,    // (B,D,H,W)
                               int NV, int D, int H, int W) {
    int i = blockIdx.x * blockDim.x + threadIdx.x;
    if (i >= NV) return;
    int b = coors[i * 4 + 0];
    int z = coors[i * 4 + 1];
    int y = coors[i * 4 + 2];
    int x = coors[i * 4 + 3];
    long long sp = (((long long)b * D + z) * H + y) * (long long)W + x;
    float* cell = dense + sp * 4;
    atomicAdd(cell + 0, vf[i * 3 + 0]);
    atomicAdd(cell + 1, vf[i * 3 + 1]);
    atomicAdd(cell + 2, vf[i * 3 + 2]);
    mask[sp] = 1.0f;                       // idempotent
}

// ------- weight prep: fp32 OIDHW -> f16 [co][ tap*CinP + ci ], zero-padded --
__global__ void wprep_kernel(const float* __restrict__ w,
                             _Float16* __restrict__ wh,
                             int Co, int Cin, int CinP, int kvol, int Kpad) {
    int i = blockIdx.x * blockDim.x + threadIdx.x;
    if (i >= Co * Kpad) return;
    int co  = i / Kpad;
    int k   = i - co * Kpad;
    int ci  = k & (CinP - 1);     // CinP is a power of two
    int tap = k / CinP;
    float v = 0.f;
    if (ci < Cin && tap < kvol)
        v = w[(size_t)co * (Cin * kvol) + (size_t)ci * kvol + tap];
    wh[i] = (_Float16)v;
}

// ---------------- generic conv block: implicit GEMM with WMMA --------------
// One wave computes a 16 (consecutive wo) x 16 (out channels) tile.
// A: 16x32 f16, element i holds K = kc + (i&7) + 8*(lane>>4) + 16*(i>>3),
//    M = lane&15.  Channels-last input => each 8-run = one v8h load (f16 in)
//    or two float4 loads (f32 in).
// B: 32x16 f16 from LDS, element i holds K = kc + (i&7) + 8*(lane>>4) + 16*(i>>3),
//    N = lane&15 (row-major [co][k] => two 16B LDS reads / lane / chunk).
// C/D: v8f, element r -> (M = r + 8*(lane>>4), N = lane&15).
// Padded K relies on zeroed weights (A x 0 == 0).
template <int KD, int KH, int KW, int CINP, bool OUT_CLAST,
          typename TIN, typename TOUT>
__global__ void __launch_bounds__(256)
conv_block_kernel(const TIN* __restrict__ in,    // (N,Din,Hin,Win,CINP)
                  const float* __restrict__ min_, // (N,Din,Hin,Win)
                  const _Float16* __restrict__ wh,
                  const float* __restrict__ g,  const float* __restrict__ bb,
                  const float* __restrict__ rm, const float* __restrict__ rv,
                  TOUT* __restrict__ out, float* __restrict__ mout,
                  int Din, int Hin, int Win,
                  int Co,  int Dout, int Hout, int Wout,
                  int sd, int sh, int sw,
                  int pd, int ph, int pw, int Kpad, int totalTiles) {
    constexpr int KHW = KH * KW;
    extern __shared__ _Float16 ldsW[];

    // ---- stage f16 weights into LDS -----------------------------------
#ifdef HAVE_TDM
    // One TDM descriptor moves the whole [Co][Kpad] f16 block (<=110592 B).
    // D# per CDNA5 ISA ch.8: count=1, type=2(image), data_size=2 (4B units),
    // one row of nD dwords (nD <= 27648 fits tile_dim0's 16 bits).
    if (threadIdx.x == 0) {
        unsigned ldsAddr = (unsigned)(size_t)ldsW;
        unsigned long long ga = (unsigned long long)(size_t)wh;
        unsigned nD = (unsigned)(Co * Kpad) >> 1;          // dwords
        u32x4 g0 = { 1u,                                   // count=1
                     ldsAddr,                              // lds_addr
                     (unsigned)ga,                         // global_addr lo
                     (unsigned)((ga >> 32) & 0x1FFFFFFu) | (2u << 30) }; // hi|type
        i32x8 g1 = { (int)(2u << 16),                      // data_size=4B
                     (int)((nD & 0xFFFFu) << 16),          // tensor_dim0 lo16
                     (int)(((nD >> 16) & 0xFFFFu) | (1u << 16)), // dim0 hi|dim1=1
                     (int)((nD & 0xFFFFu) << 16),          // tile_dim0 = nD
                     1,                                    // tile_dim1=1
                     (int)nD,                              // dim0 stride lo32
                     (int)((nD & 0xFFFFu) << 16),          // dim1 stride lo16
                     (int)((nD >> 16) & 0xFFFFu) };        // dim1 stride hi
        i32x4 gz = { 0, 0, 0, 0 };
#if __clang_major__ >= 23
        i32x8 gz8 = { 0, 0, 0, 0, 0, 0, 0, 0 };
        __builtin_amdgcn_tensor_load_to_lds(g0, g1, gz, gz, gz8, 0);
#else
        __builtin_amdgcn_tensor_load_to_lds(g0, g1, gz, gz, 0);
#endif
        __builtin_amdgcn_s_wait_tensorcnt((short)0);
    }
    __syncthreads();
#else
    int nW = Co * Kpad;
    for (int i = threadIdx.x; i < nW; i += blockDim.x) ldsW[i] = wh[i];
    __syncthreads();
#endif

    int wave = threadIdx.x >> 5;
    int lane = threadIdx.x & 31;
    int tile = blockIdx.x * (blockDim.x >> 5) + wave;
    tile = __builtin_amdgcn_readfirstlane(tile);   // wave-uniform -> SGPR
    bool active = tile < totalTiles;
    if (!active) tile = 0;

    int wblocks  = Wout >> 4;
    int coBlocks = Co   >> 4;
    int t = tile;
    int wblk = t % wblocks;  t /= wblocks;
    int hy   = t % Hout;     t /= Hout;
    int dz   = t % Dout;     t /= Dout;
    int cob  = t % coBlocks; t /= coBlocks;
    int n    = t;

    int m15 = lane & 15;
    int hi  = lane >> 4;
    int wo  = (wblk << 4) + m15;          // this lane's output position (A row)
    int co  = (cob << 4) + m15;           // this lane's output channel (B col)

    long long HWin = (long long)Hin * Win;
    const TIN* inN = in + (long long)n * Din * HWin * CINP;
    int z0 = dz * sd - pd;                 // wave-uniform
    int y0 = hy * sh - ph;                 // wave-uniform
    int x0 = wo * sw - pw;                 // per-lane

    const _Float16* wrow = ldsW + (size_t)co * Kpad;

    v8f acc = {};
    for (int kc = 0; kc < Kpad; kc += 32) {
        // ---- B fragment: two 16-byte runs (kb, kb+16) of this co's weight row ----
        int kb = kc + (hi << 3);
        v8h b0 = *(const v8h*)(wrow + kb);
        v8h b1 = *(const v8h*)(wrow + kb + 16);
        v16h bm;
#pragma unroll
        for (int i = 0; i < 8; ++i) { bm[i] = b0[i]; bm[i + 8] = b1[i]; }

        // ---- A fragment ------------------------------------------------
        v16h am;
#pragma unroll
        for (int half = 0; half < 2; ++half) {
            int Kbase = kc + (hi << 3) + (half << 4);
            int o = half << 3;
            if constexpr (CINP >= 16) {
                // 8 consecutive K = 8 contiguous channels at one tap
                int tap = Kbase / CINP;            // constant shift
                int cib = Kbase & (CINP - 1);
                int dzz = tap / KHW; int r = tap - dzz * KHW;
                int dyy = r / KW;    int dxx = r - dyy * KW;
                int zi = z0 + dzz, yi = y0 + dyy, xi = x0 + dxx;
                bool ok = (unsigned)zi < (unsigned)Din &&
                          (unsigned)yi < (unsigned)Hin &&
                          (unsigned)xi < (unsigned)Win;
                long long base = (((long long)zi * Hin + yi) * Win + xi) *
                                 CINP + cib;
                if constexpr (sizeof(TIN) == 2) {
                    v8h f = {};
                    if (ok) f = *(const v8h*)(inN + base);   // one 16B load
#pragma unroll
                    for (int j = 0; j < 8; ++j) am[o + j] = f[j];
                } else {
                    float4 f0 = make_float4(0.f, 0.f, 0.f, 0.f);
                    float4 f1 = f0;
                    if (ok) {
                        const float4* p = (const float4*)(inN + base);
                        f0 = p[0];
                        f1 = p[1];
                    }
                    am[o + 0] = (_Float16)f0.x; am[o + 1] = (_Float16)f0.y;
                    am[o + 2] = (_Float16)f0.z; am[o + 3] = (_Float16)f0.w;
                    am[o + 4] = (_Float16)f1.x; am[o + 5] = (_Float16)f1.y;
                    am[o + 6] = (_Float16)f1.z; am[o + 7] = (_Float16)f1.w;
                }
            } else {
                // CINP == 4 (layer 1, f32 dense): 8-run = two taps x 4 ch
#pragma unroll
                for (int s = 0; s < 2; ++s) {
                    int tap = (Kbase >> 2) + s;
                    int dzz = tap / KHW; int r = tap - dzz * KHW;
                    int dyy = r / KW;    int dxx = r - dyy * KW;
                    int zi = z0 + dzz, yi = y0 + dyy, xi = x0 + dxx;
                    bool ok = (unsigned)zi < (unsigned)Din &&
                              (unsigned)yi < (unsigned)Hin &&
                              (unsigned)xi < (unsigned)Win;
                    long long base = (((long long)zi * Hin + yi) * Win + xi) * 4;
                    float4 f = make_float4(0.f, 0.f, 0.f, 0.f);
                    if (ok) f = *(const float4*)((const float*)inN + base);
                    int o2 = o + (s << 2);
                    am[o2 + 0] = (_Float16)f.x; am[o2 + 1] = (_Float16)f.y;
                    am[o2 + 2] = (_Float16)f.z; am[o2 + 3] = (_Float16)f.w;
                }
            }
        }
        acc = __builtin_amdgcn_wmma_f32_16x16x32_f16(
            false, am, false, bm, (short)0, acc, false, false);
    }

    // ---- occupancy mask for position wo (every lane computes for m15) ----
    float mval = 0.f;
    {
        long long mbase = (long long)n * Din * HWin;   // mask: (N,D,H,W)
#pragma unroll
        for (int a = 0; a < KD; ++a) {
            int zi = z0 + a;
            if ((unsigned)zi >= (unsigned)Din) continue;
#pragma unroll
            for (int b2 = 0; b2 < KH; ++b2) {
                int yi = y0 + b2;
                if ((unsigned)yi >= (unsigned)Hin) continue;
#pragma unroll
                for (int c = 0; c < KW; ++c) {
                    int xi = x0 + c;
                    if ((unsigned)xi >= (unsigned)Win) continue;
                    if (min_[mbase + (long long)zi * HWin +
                             (long long)yi * Win + xi] > 0.f)
                        mval = 1.f;
                }
            }
        }
    }

    // ---- folded BN + ReLU + mask, store --------------------------------
    float inv   = g[co] * rsqrtf(rv[co] + 1e-3f);
    float shift = bb[co] - rm[co] * inv;

    if (active) {
        long long HWout = (long long)Hout * Wout;
        if constexpr (OUT_CLAST) {
            // (N,Dout,Hout,Wout,Co) f16 for the next layer
            long long pbase = ((((long long)n * Dout + dz) * Hout + hy) *
                               (long long)Wout + (wblk << 4));
            long long obase = pbase * Co + co;
#pragma unroll
            for (int r = 0; r < 8; ++r) {
                int M = r + (hi << 3);
                float mv = __shfl(mval, M, 32);
                float yv = fmaxf(acc[r] * inv + shift, 0.f) * mv;
                out[obase + (long long)M * Co] = (TOUT)yv;
            }
        } else {
            // NCDHW f32: final layer writes d_out (reshape is a pure view)
            long long obase = (((long long)n * Co + co) * Dout + dz) * HWout +
                              (long long)hy * Wout + (long long)(wblk << 4);
#pragma unroll
            for (int r = 0; r < 8; ++r) {
                int M = r + (hi << 3);
                float mv = __shfl(mval, M, 32);
                float yv = fmaxf(acc[r] * inv + shift, 0.f) * mv;
                out[obase + M] = (TOUT)yv;
            }
        }
        if (cob == 0 && lane < 16) {
            long long mo = ((long long)n * Dout + dz) * HWout +
                           (long long)hy * Wout + wo;
            mout[mo] = mval;
        }
    }
}

// ---------------------------------------------------------------------------
// Host side
// ---------------------------------------------------------------------------
// workspace layout in float units (h buffers are f16 and use half the slot)
static constexpr long long OFF_DENSE = 0;                      // 2*41*512*512*4 f32
static constexpr long long N_DENSE   = 85983232LL;
static constexpr long long OFF_MASK0 = OFF_DENSE + N_DENSE;    // 2*41*512*512 f32
static constexpr long long N_MASK0   = 21495808LL;
static constexpr long long OFF_H1    = OFF_MASK0 + N_MASK0;    // 2*21*256*256*16 f16
static constexpr long long N_H1      = 44040192LL;
static constexpr long long OFF_M1    = OFF_H1 + N_H1;          // 2*21*256*256 f32
static constexpr long long N_M1      = 2752512LL;
static constexpr long long OFF_H2    = OFF_M1 + N_M1;          // 2*11*128*128*32 f16
static constexpr long long N_H2      = 11534336LL;
static constexpr long long OFF_M2    = OFF_H2 + N_H2;          // 2*11*128*128 f32
static constexpr long long N_M2      = 720896LL;
static constexpr long long OFF_H3    = OFF_M2 + N_M2;          // 2*5*64*64*64 f16
static constexpr long long N_H3      = 2621440LL;
static constexpr long long OFF_M3    = OFF_H3 + N_H3;          // 2*5*64*64 f32
static constexpr long long N_M3      = 40960LL;
static constexpr long long OFF_M4    = OFF_M3 + N_M3;          // 2*2*64*64 f32
static constexpr long long N_M4      = 16384LL;
static constexpr long long OFF_WH    = OFF_M4 + N_M4;          // f16 weights area
// f16 offsets (in halves) inside WH area
static constexpr long long WH1 = 0;               // 16*128  = 2048
static constexpr long long WH2 = 2048;            // 32*448  = 14336
static constexpr long long WH3 = 16384;           // 64*864  = 55296
static constexpr long long WH4 = 71680;           // 64*192  = 12288

extern "C" void kernel_launch(void* const* d_in, const int* in_sizes, int n_in,
                              void* d_out, int out_size, void* d_ws, size_t ws_size,
                              hipStream_t stream) {
    const float* vf    = (const float*)d_in[0];
    const int*   coors = (const int*)d_in[1];
    const float* w1 = (const float*)d_in[3];
    const float* g1 = (const float*)d_in[4];
    const float* b1 = (const float*)d_in[5];
    const float* m1p = (const float*)d_in[6];
    const float* v1 = (const float*)d_in[7];
    const float* w2 = (const float*)d_in[8];
    const float* g2 = (const float*)d_in[9];
    const float* b2 = (const float*)d_in[10];
    const float* m2p = (const float*)d_in[11];
    const float* v2 = (const float*)d_in[12];
    const float* w3 = (const float*)d_in[13];
    const float* g3 = (const float*)d_in[14];
    const float* b3 = (const float*)d_in[15];
    const float* m3p = (const float*)d_in[16];
    const float* v3 = (const float*)d_in[17];
    const float* w4 = (const float*)d_in[18];
    const float* g4 = (const float*)d_in[19];
    const float* b4 = (const float*)d_in[20];
    const float* m4p = (const float*)d_in[21];
    const float* v4 = (const float*)d_in[22];

    float* ws = (float*)d_ws;
    float* dense = ws + OFF_DENSE;
    float* mask0 = ws + OFF_MASK0;
    _Float16* h1 = (_Float16*)(ws + OFF_H1);  float* mk1 = ws + OFF_M1;
    _Float16* h2 = (_Float16*)(ws + OFF_H2);  float* mk2 = ws + OFF_M2;
    _Float16* h3 = (_Float16*)(ws + OFF_H3);  float* mk3 = ws + OFF_M3;
    float* mk4 = ws + OFF_M4;
    _Float16* whbase = (_Float16*)(ws + OFF_WH);
    _Float16* w1h = whbase + WH1;
    _Float16* w2h = whbase + WH2;
    _Float16* w3h = whbase + WH3;
    _Float16* w4h = whbase + WH4;

    // conv3 instantiation needs 110592 B of dynamic LDS (> 64KB default cap)
    hipFuncSetAttribute(
        (const void*)conv_block_kernel<3, 3, 3, 32, true, _Float16, _Float16>,
        hipFuncAttributeMaxDynamicSharedMemorySize, 113664);

    // 1) zero dense grid + mask (contiguous region)
    zero_kernel<<<8192, 256, 0, stream>>>(dense, N_DENSE + N_MASK0);

    // 2) weight prep (f16, channel-fastest K, padded to multiple of 32)
    wprep_kernel<<<(16 * 128 + 255) / 256, 256, 0, stream>>>(w1, w1h, 16, 3,  4,  27, 128);
    wprep_kernel<<<(32 * 448 + 255) / 256, 256, 0, stream>>>(w2, w2h, 32, 16, 16, 27, 448);
    wprep_kernel<<<(64 * 864 + 255) / 256, 256, 0, stream>>>(w3, w3h, 64, 32, 32, 27, 864);
    wprep_kernel<<<(64 * 192 + 255) / 256, 256, 0, stream>>>(w4, w4h, 64, 64, 64, 3,  192);

    // 3) scatter voxels
    int NV = in_sizes[0] / 3;
    scatter_kernel<<<(NV + 255) / 256, 256, 0, stream>>>(vf, coors, dense, mask0,
                                                         NV, 41, 512, 512);

    // 4) conv blocks (8 waves / block => 8 tiles / block)
    // L1: f32 (2,41,512,512,4) -> f16 (2,21,256,256,16), k3 s2 p1
    {
        int tiles = 2 * 1 * 21 * 256 * 16;
        conv_block_kernel<3, 3, 3, 4, true, float, _Float16>
            <<<(tiles + 7) / 8, 256, 16 * 128 * 2, stream>>>(
            dense, mask0, w1h, g1, b1, m1p, v1, h1, mk1,
            41, 512, 512, 16, 21, 256, 256,
            2, 2, 2, 1, 1, 1, 128, tiles);
    }
    // L2: -> f16 (2,11,128,128,32), k3 s2 p1
    {
        int tiles = 2 * 2 * 11 * 128 * 8;
        conv_block_kernel<3, 3, 3, 16, true, _Float16, _Float16>
            <<<(tiles + 7) / 8, 256, 32 * 448 * 2, stream>>>(
            h1, mk1, w2h, g2, b2, m2p, v2, h2, mk2,
            21, 256, 256, 32, 11, 128, 128,
            2, 2, 2, 1, 1, 1, 448, tiles);
    }
    // L3: -> f16 (2,5,64,64,64), k3 s2 p(0,1,1)
    {
        int tiles = 2 * 4 * 5 * 64 * 4;
        conv_block_kernel<3, 3, 3, 32, true, _Float16, _Float16>
            <<<(tiles + 7) / 8, 256, 64 * 864 * 2, stream>>>(
            h2, mk2, w3h, g3, b3, m3p, v3, h3, mk3,
            11, 128, 128, 64, 5, 64, 64,
            2, 2, 2, 0, 1, 1, 864, tiles);
    }
    // L4: -> f32 NCDHW (2,64,2,64,64) == d_out (view), k(3,1,1) s(2,1,1) p0
    {
        int tiles = 2 * 4 * 2 * 64 * 4;
        conv_block_kernel<3, 1, 1, 64, false, _Float16, float>
            <<<(tiles + 7) / 8, 256, 64 * 192 * 2, stream>>>(
            h3, mk3, w4h, g4, b4, m4p, v4, (float*)d_out, mk4,
            5, 64, 64, 64, 2, 64, 64,
            2, 1, 1, 0, 0, 0, 192, tiles);
    }
}
